// Attention_MoE_layer_20753281974543
// MI455X (gfx1250) — matrix operand order repaired and checked by reading, not compiled, not run on hardware
//
#include <hip/hip_runtime.h>

// ---------------------------------------------------------------------------
// Types for CDNA5 (wave32) WMMA + TDM
// ---------------------------------------------------------------------------
typedef __attribute__((ext_vector_type(16))) __bf16 v16bf;
typedef __attribute__((ext_vector_type(8)))  __bf16 v8bf;
typedef __attribute__((ext_vector_type(8)))  float  v8f;
typedef __attribute__((ext_vector_type(4)))  unsigned u32x4;
typedef __attribute__((ext_vector_type(8)))  int      i32x8;
typedef __attribute__((ext_vector_type(4)))  int      i32x4;

__device__ __forceinline__ __bf16 f2bf(float f) {
    unsigned u = __builtin_bit_cast(unsigned, f);
    unsigned r = u + 0x7fffu + ((u >> 16) & 1u);
    return __builtin_bit_cast(__bf16, (unsigned short)(r >> 16));
}
__device__ __forceinline__ float bf2f(__bf16 h) {
    unsigned x = (unsigned)__builtin_bit_cast(unsigned short, h) << 16;
    return __builtin_bit_cast(float, x);
}
__device__ __forceinline__ v16bf combine16(const __bf16* lo_p, const __bf16* hi_p) {
    v8bf lo = *(const v8bf*)lo_p;
    v8bf hi = *(const v8bf*)hi_p;
    v16bf r;
#pragma unroll
    for (int t = 0; t < 8; ++t) { r[t] = lo[t]; r[t + 8] = hi[t]; }
    return r;
}
__device__ __forceinline__ void storeOut(float* p, float v)  { *p = v; }
__device__ __forceinline__ void storeOut(__bf16* p, float v) { *p = f2bf(v); }
__device__ __forceinline__ float loadOut(const float* p)  { return *p; }
__device__ __forceinline__ float loadOut(const __bf16* p) { return bf2f(*p); }

// Low 32 bits of a flat shared-aperture address == LDS byte offset.
__device__ __forceinline__ unsigned lds_off(const void* p) {
    return (unsigned)(uintptr_t)p;
}

// ---------------------------------------------------------------------------
// Tensor Data Mover: DMA a 2-D bf16 tile (tile_d0 contiguous elems per row,
// tile_d1 rows, row stride stride0 elems) from global into LDS at lds_byte.
// If pad: after each 64 B stored, skip 16 B in LDS (row 32 elems -> stride 40).
// Issued by whole wave (EXEC ignored); caller restricts to one wave and does
// s_wait_tensorcnt + barrier.
// ---------------------------------------------------------------------------
__device__ __forceinline__ void tdm_load_2d(
    unsigned lds_byte, const __bf16* gaddr,
    unsigned tile_d0, unsigned tile_d1, unsigned stride0, bool pad)
{
    const unsigned long long ga = (unsigned long long)(uintptr_t)gaddr;
    u32x4 g0;
    g0[0] = 1u;                                          // count=1, user mode
    g0[1] = lds_byte;                                    // lds_addr
    g0[2] = (unsigned)ga;                                // global_addr[31:0]
    g0[3] = (unsigned)((ga >> 32) & 0x1FFFFFFu) | (2u << 30); // addr[56:32]|type=2
    const unsigned TD = 0x01000000u;                     // huge tensor dims: no OOB
    i32x8 g1;
    g1[0] = (int)((1u << 16) |                           // data_size = 2 bytes
                  (pad ? ((1u << 20) | (3u << 22) | (3u << 25)) : 0u));
    g1[1] = (int)((TD & 0xFFFFu) << 16);                 // tensor_dim0 lo16
    g1[2] = (int)(((TD >> 16) & 0xFFFFu) | ((TD & 0xFFFFu) << 16)); // d0 hi | d1 lo
    g1[3] = (int)(((TD >> 16) & 0xFFFFu) | (tile_d0 << 16));        // d1 hi | tile_d0
    g1[4] = (int)(tile_d1 & 0xFFFFu);                    // tile_d1 | tile_d2=0
    g1[5] = (int)stride0;                                // tensor_dim0_stride lo32
    g1[6] = 0;
    g1[7] = 0;
    const i32x4 z4 = {0, 0, 0, 0};
    const i32x8 z8 = {0, 0, 0, 0, 0, 0, 0, 0};
#if __has_builtin(__builtin_amdgcn_tensor_load_to_lds)
    // clang-23 / therock lane: 6-arg form (g0, g1, g2, g3, g4, cpol)
    __builtin_amdgcn_tensor_load_to_lds(g0, g1, z4, z4, z8, 0);
#else
    asm volatile("tensor_load_to_lds %0, %1, %2, %3"
                 :: "s"(g0), "s"(g1), "s"(z4), "s"(z4) : "memory");
#endif
}

// ---------------------------------------------------------------------------
// GEMM: C[M,N] = epi(A[M,K]_bf16 * Bt[N,K]_bf16)  (Bt is pre-transposed)
// Block tile 128x128, BK=32; 256 threads = 8 waves (4x2), wave tile 32x64,
// 2x4 WMMA accumulators. A/B tiles double-buffered in LDS, fed by TDM.
// ---------------------------------------------------------------------------
#define TM 128
#define TN 128
#define TK 32
#define PK (TK + 8)

template <typename OutT, bool BIAS, bool RELU, bool SCALE, bool RESID, bool ACC>
__global__ __launch_bounds__(256) void gemm_tdm(
    const __bf16* __restrict__ A, int lda,
    const __bf16* __restrict__ Bt, int ldb,
    OutT* __restrict__ C, int ldc,
    int M, int N, int K,
    const float* __restrict__ bias,
    const float* __restrict__ resid,
    const float* __restrict__ rowscale, int rsStride)
{
    __shared__ __bf16 sA[2][TM][PK];
    __shared__ __bf16 sB[2][TN][PK];

    const int tid  = threadIdx.x;
    const int lane = tid & 31;
    const int wave = tid >> 5;
    const int wm   = wave >> 1;      // 0..3 (32-row tiles)
    const int wn   = wave & 1;       // 0..1 (64-col tiles)
    const int half = lane >> 4;
    const int lm   = lane & 15;

    const int m0 = blockIdx.y * TM;
    const int n0 = blockIdx.x * TN;
    const int nk = K / TK;

    if (wave == 0) {
        tdm_load_2d(lds_off(&sA[0][0][0]), A  + (size_t)m0 * lda, TK, TM, (unsigned)lda, true);
        tdm_load_2d(lds_off(&sB[0][0][0]), Bt + (size_t)n0 * ldb, TK, TN, (unsigned)ldb, true);
        __builtin_amdgcn_s_wait_tensorcnt(0);
    }
    __syncthreads();

    v8f acc[2][4];
#pragma unroll
    for (int i = 0; i < 2; ++i)
#pragma unroll
        for (int j = 0; j < 4; ++j)
#pragma unroll
            for (int t = 0; t < 8; ++t) acc[i][j][t] = 0.0f;

    for (int kt = 0; kt < nk; ++kt) {
        const int buf = kt & 1;
        if (wave == 0 && kt + 1 < nk) {
            const int k1 = (kt + 1) * TK;
            tdm_load_2d(lds_off(&sA[buf ^ 1][0][0]), A  + (size_t)m0 * lda + k1, TK, TM, (unsigned)lda, true);
            tdm_load_2d(lds_off(&sB[buf ^ 1][0][0]), Bt + (size_t)n0 * ldb + k1, TK, TN, (unsigned)ldb, true);
        }

        v16bf af[2], bfg[4];
#pragma unroll
        for (int i = 0; i < 2; ++i) {
            const __bf16* p = &sA[buf][wm * 32 + i * 16 + lm][half * 8];
            af[i] = combine16(p, p + 16);        // A frag: K chunks {ko, ko+16}
        }
#pragma unroll
        for (int j = 0; j < 4; ++j) {
            const __bf16* p = &sB[buf][wn * 64 + j * 16 + lm][half * 8];
            bfg[j] = combine16(p, p + 16);       // B frag: K chunks {ko, ko+16}
        }
#pragma unroll
        for (int i = 0; i < 2; ++i)
#pragma unroll
            for (int j = 0; j < 4; ++j)
                acc[i][j] = __builtin_amdgcn_wmma_f32_16x16x32_bf16(
                    false, af[i], false, bfg[j], (short)0, acc[i][j], false, false);

        if (wave == 0 && kt + 1 < nk) __builtin_amdgcn_s_wait_tensorcnt(0);
        __syncthreads();
    }

    // ---- specialized epilogue (no runtime branches) ----
#pragma unroll
    for (int i = 0; i < 2; ++i)
#pragma unroll
        for (int j = 0; j < 4; ++j) {
            const int col = n0 + wn * 64 + j * 16 + lm;
            float bv = 0.0f;
            if constexpr (BIAS) bv = bias[col];
#pragma unroll
            for (int v = 0; v < 8; ++v) {
                const int row = m0 + wm * 32 + i * 16 + half * 8 + v;
                float val = acc[i][j][v] + bv;
                if constexpr (RELU)  val = fmaxf(val, 0.0f);
                if constexpr (SCALE) val *= rowscale[(size_t)row * rsStride];
                const size_t idx = (size_t)row * ldc + col;
                if constexpr (RESID) val += resid[idx];
                if constexpr (ACC)   val += loadOut(&C[idx]);
                storeOut(&C[idx], val);
            }
        }
}

// ---------------------------------------------------------------------------
// Flash attention. Q/K bf16 [B,S,D] (head h at cols h*64..), Vt bf16 [B*H,64,S].
// 128 threads = 4 waves, each wave 16 query rows; keys in 128-wide LDS tiles
// fed by TDM. Output att bf16 [B,S,D].
// ---------------------------------------------------------------------------
__global__ __launch_bounds__(128) void attn_fwd(
    const __bf16* __restrict__ Q, const __bf16* __restrict__ K,
    const __bf16* __restrict__ Vt, __bf16* __restrict__ O)
{
    constexpr int S = 1024, D = 1024, DH = 64, KB = 128;
    __shared__ __bf16 sK [KB][DH];       // [key][dh]
    __shared__ __bf16 sVt[DH][KB];       // [dh][key]
    __shared__ __bf16 sP [4][16][40];    // per-wave P staging

    const int tid  = threadIdx.x;
    const int lane = tid & 31;
    const int wave = tid >> 5;
    const int half = lane >> 4;
    const int lm   = lane & 15;

    const int h  = blockIdx.y;
    const int b  = blockIdx.z;
    const int q0 = blockIdx.x * 64 + wave * 16;
    const size_t baseBH = (size_t)b * S * D + (size_t)h * DH;
    const size_t vtBase = ((size_t)(b * 16 + h)) * DH * S;   // Vt[b*H+h][0][0]

    v16bf aq[2];
    {
        const __bf16* qrow = Q + baseBH + (size_t)(q0 + lm) * D;
#pragma unroll
        for (int t = 0; t < 2; ++t) {
            const __bf16* p = qrow + t * 32 + half * 8;
            aq[t] = combine16(p, p + 16);
        }
    }

    v8f oacc[4];
#pragma unroll
    for (int d = 0; d < 4; ++d)
#pragma unroll
        for (int t = 0; t < 8; ++t) oacc[d][t] = 0.0f;
    float mrun[8], lrun[8];
#pragma unroll
    for (int v = 0; v < 8; ++v) { mrun[v] = -1e30f; lrun[v] = 0.0f; }

    const float scale = 0.125f;

    for (int kb0 = 0; kb0 < S; kb0 += KB) {
        if (wave == 0) {
            tdm_load_2d(lds_off(&sK[0][0]),  K  + baseBH + (size_t)kb0 * D, DH, KB, D, false);
            tdm_load_2d(lds_off(&sVt[0][0]), Vt + vtBase + kb0,             KB, DH, S, false);
            __builtin_amdgcn_s_wait_tensorcnt(0);
        }
        __syncthreads();

        for (int ks = 0; ks < KB; ks += 32) {
            v8f s0, s1;
#pragma unroll
            for (int t = 0; t < 8; ++t) { s0[t] = 0.0f; s1[t] = 0.0f; }
#pragma unroll
            for (int t = 0; t < 2; ++t) {
                const __bf16* p0 = &sK[ks + lm][t * 32 + half * 8];
                const __bf16* p1 = &sK[ks + 16 + lm][t * 32 + half * 8];
                v16bf bk0 = combine16(p0, p0 + 16);
                v16bf bk1 = combine16(p1, p1 + 16);
                s0 = __builtin_amdgcn_wmma_f32_16x16x32_bf16(false, aq[t], false, bk0, (short)0, s0, false, false);
                s1 = __builtin_amdgcn_wmma_f32_16x16x32_bf16(false, aq[t], false, bk1, (short)0, s1, false, false);
            }

            float p0a[8], p1a[8], corr[8];
#pragma unroll
            for (int v = 0; v < 8; ++v) {
                float a0 = s0[v] * scale;
                float a1 = s1[v] * scale;
                float mx = fmaxf(a0, a1);
#pragma unroll
                for (int off = 1; off < 16; off <<= 1)
                    mx = fmaxf(mx, __shfl_xor(mx, off, 32));
                float mnew = fmaxf(mrun[v], mx);
                float c    = __expf(mrun[v] - mnew);
                float e0   = __expf(a0 - mnew);
                float e1   = __expf(a1 - mnew);
                float rs   = e0 + e1;
#pragma unroll
                for (int off = 1; off < 16; off <<= 1)
                    rs += __shfl_xor(rs, off, 32);
                lrun[v] = lrun[v] * c + rs;
                mrun[v] = mnew;
                corr[v] = c; p0a[v] = e0; p1a[v] = e1;
            }
#pragma unroll
            for (int d = 0; d < 4; ++d)
#pragma unroll
                for (int v = 0; v < 8; ++v) oacc[d][v] *= corr[v];

#pragma unroll
            for (int v = 0; v < 8; ++v) {
                sP[wave][half * 8 + v][lm]      = f2bf(p0a[v]);
                sP[wave][half * 8 + v][16 + lm] = f2bf(p1a[v]);
            }
            __syncthreads();

            const __bf16* pp = &sP[wave][lm][half * 8];
            v16bf ap = combine16(pp, pp + 16);
#pragma unroll
            for (int d = 0; d < 4; ++d) {
                const __bf16* pv = &sVt[d * 16 + lm][ks + half * 8];
                v16bf bv = combine16(pv, pv + 16);
                oacc[d] = __builtin_amdgcn_wmma_f32_16x16x32_bf16(
                    false, ap, false, bv, (short)0, oacc[d], false, false);
            }
        }
        __syncthreads();
    }

#pragma unroll
    for (int d = 0; d < 4; ++d)
#pragma unroll
        for (int v = 0; v < 8; ++v) {
            const int row = q0 + half * 8 + v;
            O[baseBH + (size_t)row * D + d * 16 + lm] = f2bf(oacc[d][v] / lrun[v]);
        }
}

// ---------------------------------------------------------------------------
// RMSNorm: fp32 in, bf16 out (feeds GEMM A path)
// ---------------------------------------------------------------------------
__global__ __launch_bounds__(256) void rmsnorm_kernel(
    const float* __restrict__ x, const float* __restrict__ g,
    __bf16* __restrict__ y, int D)
{
    __shared__ float red[8];
    const int row = blockIdx.x;
    const float* xr = x + (size_t)row * D;
    float s = 0.0f;
    for (int d = threadIdx.x; d < D; d += 256) { float v = xr[d]; s += v * v; }
#pragma unroll
    for (int off = 1; off < 32; off <<= 1) s += __shfl_xor(s, off, 32);
    if ((threadIdx.x & 31) == 0) red[threadIdx.x >> 5] = s;
    __syncthreads();
    if (threadIdx.x < 8) {
        float t = red[threadIdx.x];
#pragma unroll
        for (int off = 1; off < 8; off <<= 1) t += __shfl_xor(t, off, 8);
        if (threadIdx.x == 0) red[0] = t;
    }
    __syncthreads();
    const float rs = rsqrtf(red[0] / (float)D + 1e-6f);
    for (int d = threadIdx.x; d < D; d += 256)
        y[(size_t)row * D + d] = f2bf(xr[d] * rs * g[d]);
}

// ---------------------------------------------------------------------------
// Weight prep: W[Kd][Nd] fp32 -> Wt[Nd][Kd] bf16 (batched over blockIdx.z)
// ---------------------------------------------------------------------------
__global__ __launch_bounds__(256) void wtrans_kernel(
    const float* __restrict__ W, __bf16* __restrict__ Wt, int Kd, int Nd)
{
    __shared__ float t[32][33];
    const float*  Wz  = W  + (size_t)blockIdx.z * Kd * Nd;
    __bf16*       Wtz = Wt + (size_t)blockIdx.z * Kd * Nd;
    const int k0 = blockIdx.y * 32, n0 = blockIdx.x * 32;
    const int tx = threadIdx.x & 31, ty = threadIdx.x >> 5;
#pragma unroll
    for (int r = 0; r < 4; ++r)
        t[ty + r * 8][tx] = Wz[(size_t)(k0 + ty + r * 8) * Nd + n0 + tx];
    __syncthreads();
#pragma unroll
    for (int r = 0; r < 4; ++r)
        Wtz[(size_t)(n0 + ty + r * 8) * Kd + k0 + tx] = f2bf(t[tx][ty + r * 8]);
}

// ---------------------------------------------------------------------------
// V transpose per head: V bf16 [B,S,D] -> Vt bf16 [B*H, 64, S]
// ---------------------------------------------------------------------------
__global__ __launch_bounds__(256) void vtrans_kernel(
    const __bf16* __restrict__ V, __bf16* __restrict__ Vt)
{
    __shared__ __bf16 t[32][33];
    const int bh = blockIdx.z, b = bh >> 4, h = bh & 15;
    const int s0 = blockIdx.x * 32, d0 = blockIdx.y * 32;
    const int tx = threadIdx.x & 31, ty = threadIdx.x >> 5;
#pragma unroll
    for (int r = 0; r < 4; ++r)
        t[ty + r * 8][tx] =
            V[((size_t)b * 1024 + s0 + ty + r * 8) * 1024 + h * 64 + d0 + tx];
    __syncthreads();
#pragma unroll
    for (int r = 0; r < 4; ++r)
        Vt[((size_t)bh * 64 + d0 + ty + r * 8) * 1024 + s0 + tx] = t[tx][ty + r * 8];
}

// ---------------------------------------------------------------------------
// Gating: gate[row,:] = softmax(xn[row,:] @ Wg), one wave per row
// ---------------------------------------------------------------------------
__global__ __launch_bounds__(32) void gate_kernel(
    const __bf16* __restrict__ xn, const float* __restrict__ Wg,
    float* __restrict__ gate)
{
    const int row = blockIdx.x;
    const __bf16* xr = xn + (size_t)row * 1024;
    float acc[8];
#pragma unroll
    for (int e = 0; e < 8; ++e) acc[e] = 0.0f;
    for (int d = threadIdx.x; d < 1024; d += 32) {
        const float xv = bf2f(xr[d]);
        const float* w = Wg + (size_t)d * 8;
#pragma unroll
        for (int e = 0; e < 8; ++e) acc[e] += xv * w[e];
    }
#pragma unroll
    for (int e = 0; e < 8; ++e)
#pragma unroll
        for (int off = 1; off < 32; off <<= 1)
            acc[e] += __shfl_xor(acc[e], off, 32);
    float mx = acc[0];
#pragma unroll
    for (int e = 1; e < 8; ++e) mx = fmaxf(mx, acc[e]);
    float p[8], sum = 0.0f;
#pragma unroll
    for (int e = 0; e < 8; ++e) { p[e] = __expf(acc[e] - mx); sum += p[e]; }
    const float inv = 1.0f / sum;
    if (threadIdx.x < 8) gate[(size_t)row * 8 + threadIdx.x] = p[threadIdx.x] * inv;
}

// ---------------------------------------------------------------------------
// Launch sequence
// ---------------------------------------------------------------------------
extern "C" void kernel_launch(void* const* d_in, const int* in_sizes, int n_in,
                              void* d_out, int out_size, void* d_ws, size_t ws_size,
                              hipStream_t stream) {
    (void)in_sizes; (void)n_in; (void)out_size; (void)ws_size;
    const float* x  = (const float*)d_in[0];
    const float* g1 = (const float*)d_in[1];
    const float* Wq = (const float*)d_in[2];
    const float* Wk = (const float*)d_in[3];
    const float* Wv = (const float*)d_in[4];
    const float* Wo = (const float*)d_in[5];
    const float* g2 = (const float*)d_in[6];
    const float* Wg = (const float*)d_in[7];
    const float* W1 = (const float*)d_in[8];
    const float* b1 = (const float*)d_in[9];
    const float* W2 = (const float*)d_in[10];
    const float* b2 = (const float*)d_in[11];
    float* out = (float*)d_out;

    constexpr int Bc = 8, Sc = 1024, Dc = 1024, Hc = 16, Fc = 4096, Ec = 8;
    constexpr size_t BS = (size_t)Bc * Sc;      // 8192
    constexpr int CHUNK = 2048;

    char* ws = (char*)d_ws;
    size_t off = 0;
    auto alloc = [&](size_t bytes) -> void* {
        void* p = ws + off;
        off += (bytes + 255) & ~(size_t)255;
        return p;
    };
    __bf16* Wqt = (__bf16*)alloc((size_t)Dc * Dc * 2);
    __bf16* Wkt = (__bf16*)alloc((size_t)Dc * Dc * 2);
    __bf16* Wvt = (__bf16*)alloc((size_t)Dc * Dc * 2);
    __bf16* Wot = (__bf16*)alloc((size_t)Dc * Dc * 2);
    __bf16* W1t = (__bf16*)alloc((size_t)Ec * Dc * Fc * 2);
    __bf16* W2t = (__bf16*)alloc((size_t)Ec * Fc * Dc * 2);
    __bf16* xn  = (__bf16*)alloc(BS * Dc * 2);
    __bf16* Qb  = (__bf16*)alloc(BS * Dc * 2);
    __bf16* Kb  = (__bf16*)alloc(BS * Dc * 2);
    __bf16* Vb  = (__bf16*)alloc(BS * Dc * 2);
    __bf16* Vt  = (__bf16*)alloc(BS * Dc * 2);
    __bf16* att = (__bf16*)alloc(BS * Dc * 2);
    float*  x1  = (float*)alloc(BS * Dc * 4);
    float*  gate= (float*)alloc(BS * Ec * 4);
    __bf16* hbuf= (__bf16*)alloc((size_t)CHUNK * Fc * 2);

    // 0) weight prep: fp32 -> bf16, transposed to [N][K]
    wtrans_kernel<<<dim3(Dc/32, Dc/32, 1), 256, 0, stream>>>(Wq, Wqt, Dc, Dc);
    wtrans_kernel<<<dim3(Dc/32, Dc/32, 1), 256, 0, stream>>>(Wk, Wkt, Dc, Dc);
    wtrans_kernel<<<dim3(Dc/32, Dc/32, 1), 256, 0, stream>>>(Wv, Wvt, Dc, Dc);
    wtrans_kernel<<<dim3(Dc/32, Dc/32, 1), 256, 0, stream>>>(Wo, Wot, Dc, Dc);
    wtrans_kernel<<<dim3(Fc/32, Dc/32, Ec), 256, 0, stream>>>(W1, W1t, Dc, Fc);
    wtrans_kernel<<<dim3(Dc/32, Fc/32, Ec), 256, 0, stream>>>(W2, W2t, Fc, Dc);

    // 1) xn = rmsnorm(x, g1)   (bf16)
    rmsnorm_kernel<<<(unsigned)BS, 256, 0, stream>>>(x, g1, xn, Dc);

    // 2) Q/K/V projections (bf16 out)
    dim3 gProj(Dc / TN, (unsigned)(BS / TM));
    gemm_tdm<__bf16,false,false,false,false,false><<<gProj, 256, 0, stream>>>(
        xn, Dc, Wqt, Dc, Qb, Dc, (int)BS, Dc, Dc, nullptr, nullptr, nullptr, 0);
    gemm_tdm<__bf16,false,false,false,false,false><<<gProj, 256, 0, stream>>>(
        xn, Dc, Wkt, Dc, Kb, Dc, (int)BS, Dc, Dc, nullptr, nullptr, nullptr, 0);
    gemm_tdm<__bf16,false,false,false,false,false><<<gProj, 256, 0, stream>>>(
        xn, Dc, Wvt, Dc, Vb, Dc, (int)BS, Dc, Dc, nullptr, nullptr, nullptr, 0);

    // 3) per-head V transpose, then flash attention (bf16 out)
    vtrans_kernel<<<dim3(Sc/32, 2, Bc*Hc), 256, 0, stream>>>(Vb, Vt);
    attn_fwd<<<dim3(Sc/64, Hc, Bc), 128, 0, stream>>>(Qb, Kb, Vt, att);

    // 4) x1 = att @ Wo + x   (fp32 out)
    gemm_tdm<float,false,false,false,true,false><<<gProj, 256, 0, stream>>>(
        att, Dc, Wot, Dc, x1, Dc, (int)BS, Dc, Dc, nullptr, x, nullptr, 0);

    // 5) xn = rmsnorm(x1, g2); gate = softmax(xn @ Wg)
    rmsnorm_kernel<<<(unsigned)BS, 256, 0, stream>>>(x1, g2, xn, Dc);
    gate_kernel<<<(unsigned)BS, 32, 0, stream>>>(xn, Wg, gate);

    // 6) dense MoE: out = x1 + sum_e gate_e * (relu(xn W1_e + b1_e) W2_e + b2_e)
    for (int e = 0; e < Ec; ++e) {
        for (int c = 0; c < (int)(BS / CHUNK); ++c) {
            const size_t r0 = (size_t)c * CHUNK;
            gemm_tdm<__bf16,true,true,false,false,false>
                <<<dim3(Fc/TN, CHUNK/TM), 256, 0, stream>>>(
                xn + r0 * Dc, Dc, W1t + (size_t)e * Dc * Fc, Dc, hbuf, Fc,
                CHUNK, Fc, Dc, b1 + (size_t)e * Fc, nullptr, nullptr, 0);
            if (e == 0) {
                gemm_tdm<float,true,false,true,true,false>
                    <<<dim3(Dc/TN, CHUNK/TM), 256, 0, stream>>>(
                    hbuf, Fc, W2t + (size_t)e * Fc * Dc, Fc, out + r0 * Dc, Dc,
                    CHUNK, Dc, Fc, b2 + (size_t)e * Dc,
                    x1 + r0 * Dc, gate + r0 * Ec + e, Ec);
            } else {
                gemm_tdm<float,true,false,true,false,true>
                    <<<dim3(Dc/TN, CHUNK/TM), 256, 0, stream>>>(
                    hbuf, Fc, W2t + (size_t)e * Fc * Dc, Fc, out + r0 * Dc, Dc,
                    CHUNK, Dc, Fc, b2 + (size_t)e * Dc,
                    nullptr, gate + r0 * Ec + e, Ec);
            }
        }
    }
}
